// Propagation_67448166417075
// MI455X (gfx1250) — matrix operand order repaired and checked by reading, not compiled, not run on hardware
//
#include <hip/hip_runtime.h>

// ---------- types ----------
typedef unsigned short u16;
typedef unsigned int   u32;
typedef __bf16 bf16_t;
typedef bf16_t v16bf __attribute__((ext_vector_type(16)));
typedef float  v8f   __attribute__((ext_vector_type(8)));
typedef u32    u32x4 __attribute__((ext_vector_type(4)));

union Frag { u32x4 q[2]; v16bf v; };

// ---------- helpers ----------
__device__ __forceinline__ u16 f2bf(float f) {
  u32 u = __float_as_uint(f);
  u32 r = (u + 0x7fffu + ((u >> 16) & 1u)) >> 16;   // round-to-nearest-even
  return (u16)r;
}
__device__ __forceinline__ float bf2f(u16 b) {
  return __uint_as_float(((u32)b) << 16);
}
// order-preserving float <-> uint (for atomicMax on f32 values)
__device__ __forceinline__ u32 f2ord(float f) {
  u32 u = __float_as_uint(f);
  return (u & 0x80000000u) ? ~u : (u | 0x80000000u);
}
__device__ __forceinline__ float ord2f(u32 u) {
  u32 b = (u & 0x80000000u) ? (u & 0x7fffffffu) : ~u;
  return __uint_as_float(b);
}
__device__ __forceinline__ v16bf load_frag2(const u16* p0, const u16* p1) {
  Frag f;
  f.q[0] = *(const u32x4*)p0;
  f.q[1] = *(const u32x4*)p1;
  return f.v;
}

// ---------- weight convert + transpose: W[K,Nout] f32 -> Wt[Nout,K] bf16 ----------
__global__ void k_cvt_transpose(const float* __restrict__ W, u16* __restrict__ Wt,
                                int K, int Nout) {
  int idx = blockIdx.x * blockDim.x + threadIdx.x;
  if (idx >= K * Nout) return;
  int k = idx / Nout, n = idx % Nout;
  Wt[(u32)n * K + k] = f2bf(W[idx]);
}

// ---------- init: h = data (f32), hb = bf16(data), datab = bf16(data) ----------
__global__ void k_init(const float* __restrict__ data, float* __restrict__ h,
                       u16* __restrict__ hb, u16* __restrict__ datab, int n) {
  int idx = blockIdx.x * blockDim.x + threadIdx.x;
  if (idx >= n) return;
  float v = data[idx];
  h[idx] = v;
  u16 b = f2bf(v);
  hb[idx] = b;
  datab[idx] = b;
}

__global__ void k_cvt_bf16(const float* __restrict__ x, u16* __restrict__ y, int n) {
  int idx = blockIdx.x * blockDim.x + threadIdx.x;
  if (idx < n) y[idx] = f2bf(x[idx]);
}

__global__ void k_zero(float* __restrict__ msg, int nMsg,
                       float* __restrict__ den, u32* __restrict__ segmax, int n) {
  int idx = blockIdx.x * blockDim.x + threadIdx.x;
  if (idx < nMsg) msg[idx] = 0.0f;
  if (idx < n) { den[idx] = 0.0f; segmax[idx] = 0u; }
}

// ---------- WMMA GEMM (compile-time K): out = soft_sign(A @ Wt^T + bias) ----------
// A: [M, K1] bf16 row-major, optional A2: [M, K2] concatenated along K.
// Wt: [Nout, K1+K2] bf16 (pre-transposed weights).
// One wave per 16x16 output tile; M, Nout multiples of 16; K1,K2 multiples of 32.
template<int K1, int K2>
__global__ void k_gemm_t(const u16* __restrict__ A, const u16* __restrict__ A2,
                         const u16* __restrict__ Wt, const float* __restrict__ bias,
                         float* __restrict__ outF, u16* __restrict__ outB,
                         int M, int Nout) {
  constexpr int K = K1 + K2;
  int wave = (int)((blockIdx.x * blockDim.x + threadIdx.x) >> 5);
  int nT = Nout >> 4;
  int total = (M >> 4) * nT;
  if (wave >= total) return;                 // wave-uniform: EXEC stays all-1s
  int m0 = (wave / nT) << 4;
  int n0 = (wave % nT) << 4;
  int lane = threadIdx.x & 31;
  int lhi = lane >> 4;                       // 0 or 1 (lane half)
  int l16 = lane & 15;
  int aLo = lhi * 8;                         // A K-offset pattern (ISA 7.12.2)
  int bK  = lhi * 8;                         // B K-offset pattern

  const u16* aBase1 = A + (u32)(m0 + l16) * K1 + aLo;
  const u16* aBase2 = (K2 > 0) ? (A2 + (u32)(m0 + l16) * K2 + aLo) : A;
  const u16* bBase  = Wt + (u32)(n0 + l16) * K + bK;

  v8f acc;
  #pragma unroll
  for (int r = 0; r < 8; ++r) acc[r] = 0.0f;

  #pragma unroll
  for (int k0 = 0; k0 < K; k0 += 32) {
    const u16* ap;
    if constexpr (K2 == 0) {
      ap = aBase1 + k0;
    } else {
      ap = (k0 < K1) ? (aBase1 + k0) : (aBase2 + (k0 - K1));  // k0 is a constant
    }
    v16bf av = load_frag2(ap, ap + 16);
    const u16* bp = bBase + k0;
    v16bf bv = load_frag2(bp, bp + 16);
    acc = __builtin_amdgcn_wmma_f32_16x16x32_bf16(false, av, false, bv,
                                                  (short)0, acc, false, false);
  }

  int col = n0 + l16;
  float bb = bias[col];
  #pragma unroll
  for (int r = 0; r < 8; ++r) {
    int row = m0 + r + 8 * lhi;
    float v = acc[r] + bb;
    v = v / (1.0f + fabsf(v));               // soft_sign
    if (outF) outF[(u32)row * Nout + col] = v;
    if (outB) outB[(u32)row * Nout + col] = f2bf(v);
  }
}

// ---------- gate layer 3: g = g2 @ gw3 + gb3, K=32, Nout=1 ----------
__global__ void k_gate3(const u16* __restrict__ g2b, const float* __restrict__ gw3,
                        const float* __restrict__ gb3, float* __restrict__ g, int M) {
  int i = blockIdx.x * blockDim.x + threadIdx.x;
  if (i >= M) return;
  float s = gb3[0];
  #pragma unroll
  for (int j = 0; j < 32; ++j) s += bf2f(g2b[(u32)i * 32 + j]) * gw3[j];
  g[i] = s;
}

// ---------- edge pass 1: segment max ----------
__global__ void k_edge_max(const int* __restrict__ src, const int* __restrict__ dst,
                           const float* __restrict__ g, u32* __restrict__ segmax, int E) {
  int e = blockIdx.x * blockDim.x + threadIdx.x;
  if (e >= E) return;
  atomicMax(&segmax[dst[e]], f2ord(g[src[e]]));
}

// ---------- edge pass 2: a = exp(ge - max), den += a ----------
__global__ void k_edge_expsum(const int* __restrict__ src, const int* __restrict__ dst,
                              const float* __restrict__ g, const u32* __restrict__ segmax,
                              float* __restrict__ den, float* __restrict__ a_e, int E) {
  int e = blockIdx.x * blockDim.x + threadIdx.x;
  if (e >= E) return;
  int d = dst[e];
  float a = expf(g[src[e]] - ord2f(segmax[d]));
  a_e[e] = a;
  atomicAdd(&den[d], a);
}

// ---------- edge pass 3: msg[dst] += (a/den[dst]) * h[src]  (thread = (edge,feat)) ----------
__global__ void k_edge_scatter(const int* __restrict__ src, const int* __restrict__ dst,
                               const float* __restrict__ a_e, const float* __restrict__ den,
                               const float* __restrict__ h, float* __restrict__ msg,
                               long total) {
  long idx = (long)blockIdx.x * blockDim.x + threadIdx.x;
  if (idx >= total) return;
  int e = (int)(idx >> 7);
  int f = (int)(idx & 127);
  int d = dst[e];
  float alpha = a_e[e] / den[d];
  atomicAdd(&msg[(u32)d * 128 + f], alpha * h[(u32)src[e] * 128 + f]);
}

// ---------- fused GRU cell: one wave per 16x16 tile of h_new, 6 WMMA accumulators ----------
__global__ void k_gru(const u16* __restrict__ msgb, const u16* __restrict__ hbc,
                      const float* __restrict__ hc,
                      const u16* __restrict__ w_iht, const u16* __restrict__ w_hht,
                      const float* __restrict__ b_ih, const float* __restrict__ b_hh,
                      float* __restrict__ hN, u16* __restrict__ hbN, int M) {
  int wave = (int)((blockIdx.x * blockDim.x + threadIdx.x) >> 5);
  const int nT = 8;                           // 128/16 column tiles
  int total = (M >> 4) * nT;
  if (wave >= total) return;
  int m0 = (wave / nT) << 4;
  int n0 = (wave % nT) << 4;
  int lane = threadIdx.x & 31;
  int lhi = lane >> 4, l16 = lane & 15;
  int aLo = lhi * 8, bK = lhi * 8;

  const u16* mBase = msgb + (u32)(m0 + l16) * 128 + aLo;
  const u16* hBase = hbc  + (u32)(m0 + l16) * 128 + aLo;
  const u16* iBase = w_iht + (u32)(n0 + l16) * 128 + bK;  // + gsel*128*128
  const u16* hwBase = w_hht + (u32)(n0 + l16) * 128 + bK;

  v8f acc[6];
  #pragma unroll
  for (int i = 0; i < 6; ++i)
    #pragma unroll
    for (int r = 0; r < 8; ++r) acc[i][r] = 0.0f;

  #pragma unroll
  for (int k0 = 0; k0 < 128; k0 += 32) {
    v16bf aM = load_frag2(mBase + k0, mBase + k0 + 16);
    v16bf aH = load_frag2(hBase + k0, hBase + k0 + 16);
    #pragma unroll
    for (int gsel = 0; gsel < 3; ++gsel) {
      const u16* bi = iBase + (u32)gsel * (128 * 128) + k0;
      v16bf bI = load_frag2(bi, bi + 16);
      acc[gsel] = __builtin_amdgcn_wmma_f32_16x16x32_bf16(false, aM, false, bI,
                                                          (short)0, acc[gsel], false, false);
      const u16* bh = hwBase + (u32)gsel * (128 * 128) + k0;
      v16bf bH = load_frag2(bh, bh + 16);
      acc[3 + gsel] = __builtin_amdgcn_wmma_f32_16x16x32_bf16(false, aH, false, bH,
                                                              (short)0, acc[3 + gsel], false, false);
    }
  }

  int col = n0 + l16;
  float bir = b_ih[col], biz = b_ih[col + 128], bin_ = b_ih[col + 256];
  float bhr = b_hh[col], bhz = b_hh[col + 128], bhn = b_hh[col + 256];
  #pragma unroll
  for (int r = 0; r < 8; ++r) {
    u32 ofs = (u32)(m0 + r + 8 * lhi) * 128 + col;
    float ir = acc[0][r] + bir, iz = acc[1][r] + biz, inn = acc[2][r] + bin_;
    float hr = acc[3][r] + bhr, hz = acc[4][r] + bhz, hnn = acc[5][r] + bhn;
    float rg = 1.0f / (1.0f + expf(-(ir + hr)));
    float zg = 1.0f / (1.0f + expf(-(iz + hz)));
    float ng = tanhf(inn + rg * hnn);
    float ho = hc[ofs];
    float hv = (1.0f - zg) * ng + zg * ho;
    hN[ofs] = hv;
    hbN[ofs] = f2bf(hv);
  }
}

// ---------- final: out = softmax_row(ai) * aj, wave per row of 128 ----------
__global__ void k_softmax_mul(float* __restrict__ ai, const float* __restrict__ aj, int M) {
  int wave = (int)((blockIdx.x * blockDim.x + threadIdx.x) >> 5);
  int lane = threadIdx.x & 31;
  if (wave >= M) return;
  float v[4], e[4];
  float m = -3.402823e38f;
  #pragma unroll
  for (int j = 0; j < 4; ++j) {
    v[j] = ai[(u32)wave * 128 + lane + 32 * j];
    m = fmaxf(m, v[j]);
  }
  #pragma unroll
  for (int s = 16; s > 0; s >>= 1) m = fmaxf(m, __shfl_xor(m, s, 32));
  float sum = 0.0f;
  #pragma unroll
  for (int j = 0; j < 4; ++j) { e[j] = expf(v[j] - m); sum += e[j]; }
  #pragma unroll
  for (int s = 16; s > 0; s >>= 1) sum += __shfl_xor(sum, s, 32);
  float inv = 1.0f / sum;
  #pragma unroll
  for (int j = 0; j < 4; ++j)
    ai[(u32)wave * 128 + lane + 32 * j] = e[j] * inv * aj[(u32)wave * 128 + lane + 32 * j];
}

// =====================================================================
extern "C" void kernel_launch(void* const* d_in, const int* in_sizes, int n_in,
                              void* d_out, int out_size, void* d_ws, size_t ws_size,
                              hipStream_t stream) {
  const float* data  = (const float*)d_in[0];
  const int*   eidx  = (const int*)d_in[1];
  // d_in[2] = edge_feat (unused by reference)
  const float* gw1 = (const float*)d_in[3];  const float* gb1 = (const float*)d_in[4];
  const float* gw2 = (const float*)d_in[5];  const float* gb2 = (const float*)d_in[6];
  const float* gw3 = (const float*)d_in[7];  const float* gb3 = (const float*)d_in[8];
  const float* w_ih = (const float*)d_in[9];  const float* b_ih = (const float*)d_in[10];
  const float* w_hh = (const float*)d_in[11]; const float* b_hh = (const float*)d_in[12];
  const float* wi1 = (const float*)d_in[13]; const float* bi1 = (const float*)d_in[14];
  const float* wi2 = (const float*)d_in[15]; const float* bi2 = (const float*)d_in[16];
  const float* wj1 = (const float*)d_in[17]; const float* bj1 = (const float*)d_in[18];

  const int F = 128;
  const int N = in_sizes[0] / F;        // 20000 (multiple of 16)
  const int E = in_sizes[1] / 2;        // 640000
  const int* src = eidx;
  const int* dst = eidx + E;

  // ---- workspace carve-out (all 256B aligned) ----
  size_t off = 0;
  auto alloc = [&](size_t bytes) -> void* {
    void* p = (char*)d_ws + off;
    off += (bytes + 255) & ~(size_t)255;
    return p;
  };
  float* hX    = (float*)alloc((size_t)N * F * 4);   // h / msg ping-pong
  float* hY    = (float*)alloc((size_t)N * F * 4);
  u16*   hbA   = (u16*)  alloc((size_t)N * F * 2);   // bf16 h ping-pong
  u16*   hbB   = (u16*)  alloc((size_t)N * F * 2);
  u16*   datab = (u16*)  alloc((size_t)N * F * 2);
  u16*   msgb  = (u16*)  alloc((size_t)N * F * 2);   // reused as t1b in finale
  float* gbuf  = (float*)alloc((size_t)N * 4);
  u16*   g1b   = (u16*)  alloc((size_t)N * 64 * 2);
  u16*   g2b   = (u16*)  alloc((size_t)N * 32 * 2);
  u32*   segmx = (u32*)  alloc((size_t)N * 4);
  float* den   = (float*)alloc((size_t)N * 4);
  float* a_e   = (float*)alloc((size_t)E * 4);
  u16* gw1t  = (u16*)alloc(64  * 128 * 2);
  u16* gw2t  = (u16*)alloc(32  * 64  * 2);
  u16* w_iht = (u16*)alloc(384 * 128 * 2);
  u16* w_hht = (u16*)alloc(384 * 128 * 2);
  u16* wi1t  = (u16*)alloc(128 * 256 * 2);
  u16* wi2t  = (u16*)alloc(128 * 128 * 2);
  u16* wj1t  = (u16*)alloc(128 * 128 * 2);

  auto nblk = [](long t, int b) { return (unsigned)((t + b - 1) / b); };
  auto cvtT = [&](const float* W, u16* Wt, int K, int Nout) {
    k_cvt_transpose<<<nblk((long)K * Nout, 256), 256, 0, stream>>>(W, Wt, K, Nout);
  };
  auto gemmGrid = [&](int M, int Nout) {
    long waves = (long)(M / 16) * (Nout / 16);
    return nblk(waves * 32, 256);
  };

  // ---- weight conversion (once per call) ----
  cvtT(gw1, gw1t, 128, 64);
  cvtT(gw2, gw2t, 64, 32);
  cvtT(w_ih, w_iht, 128, 384);
  cvtT(w_hh, w_hht, 128, 384);
  cvtT(wi1, wi1t, 256, 128);
  cvtT(wi2, wi2t, 128, 128);
  cvtT(wj1, wj1t, 128, 128);

  // ---- init h = data ----
  k_init<<<nblk((long)N * F, 256), 256, 0, stream>>>(data, hX, hbA, datab, N * F);

  float* hcur = hX;  float* hnxt = hY;
  u16*   hbcur = hbA; u16*  hbnxt = hbB;

  for (int it = 0; it < 4; ++it) {
    // gate_net(h): 128 -> 64 -> 32 -> 1
    k_gemm_t<128, 0><<<gemmGrid(N, 64), 256, 0, stream>>>(
        hbcur, nullptr, gw1t, gb1, nullptr, g1b, N, 64);
    k_gemm_t<64, 0><<<gemmGrid(N, 32), 256, 0, stream>>>(
        g1b, nullptr, gw2t, gb2, nullptr, g2b, N, 32);
    k_gate3<<<nblk(N, 256), 256, 0, stream>>>(g2b, gw3, gb3, gbuf, N);

    // attentional aggregation into hnxt (acts as msg buffer)
    k_zero<<<nblk((long)N * F, 256), 256, 0, stream>>>(hnxt, N * F, den, segmx, N);
    k_edge_max<<<nblk(E, 256), 256, 0, stream>>>(src, dst, gbuf, segmx, E);
    k_edge_expsum<<<nblk(E, 256), 256, 0, stream>>>(src, dst, gbuf, segmx, den, a_e, E);
    k_edge_scatter<<<nblk((long)E * F, 256), 256, 0, stream>>>(src, dst, a_e, den,
                                                               hcur, hnxt, (long)E * F);
    k_cvt_bf16<<<nblk((long)N * F, 256), 256, 0, stream>>>(hnxt, msgb, N * F);

    // GRU: reads msgb/hbcur/hcur, overwrites hnxt with h_new (+ hbnxt)
    {
      long waves = (long)(N / 16) * 8;
      k_gru<<<nblk(waves * 32, 256), 256, 0, stream>>>(msgb, hbcur, hcur, w_iht, w_hht,
                                                       b_ih, b_hh, hnxt, hbnxt, N);
    }
    { float* t = hcur; hcur = hnxt; hnxt = t; }
    { u16* t = hbcur; hbcur = hbnxt; hbnxt = t; }
  }

  // ---- final attention: ai = ss(ss([h|data]@wi1+bi1)@wi2+bi2); aj = ss(data@wj1+bj1)
  u16*   t1b = msgb;       // free after iterations
  float* ajf = hnxt;       // free buffer (final h lives in hcur)
  k_gemm_t<128, 128><<<gemmGrid(N, 128), 256, 0, stream>>>(
      hbcur, datab, wi1t, bi1, nullptr, t1b, N, 128);
  k_gemm_t<128, 0><<<gemmGrid(N, 128), 256, 0, stream>>>(
      t1b, nullptr, wi2t, bi2, (float*)d_out, nullptr, N, 128);
  k_gemm_t<128, 0><<<gemmGrid(N, 128), 256, 0, stream>>>(
      datab, nullptr, wj1t, bj1, ajf, nullptr, N, 128);

  // out = softmax(ai, axis=1) * aj   (in-place on d_out)
  k_softmax_mul<<<nblk((long)N * 32, 256), 256, 0, stream>>>((float*)d_out, ajf, N);
}
